// LlamaAttention_83167746719895
// MI455X (gfx1250) — matrix operand order repaired and checked by reading, not compiled, not run on hardware
//
#include <hip/hip_runtime.h>
#include <hip/hip_bf16.h>

// ---------------------------------------------------------------------------
// LlamaAttention for MI455X (gfx1250): bf16 WMMA GEMMs (async-LDS staged,
// double buffered) + flash-style WMMA attention.
// ---------------------------------------------------------------------------

#define HID_C 2048
#define NH_C  16
#define HD_C  128
#define L_C   2048
#define B_C   2
#define TOK_C (B_C * L_C)   // 4096 tokens
#define NEGV  (-1.0e9f)

typedef __bf16 bf16_t;
typedef __attribute__((ext_vector_type(16))) __bf16 v16bf;
typedef __attribute__((ext_vector_type(8)))  __bf16 v8bf;
typedef __attribute__((ext_vector_type(4)))  __bf16 v4bf;
typedef __attribute__((ext_vector_type(8)))  float  v8f;
typedef __attribute__((ext_vector_type(4)))  float  v4f;

static __device__ __forceinline__ v8f vzero8() {
  v8f z;
#pragma unroll
  for (int i = 0; i < 8; ++i) z[i] = 0.0f;
  return z;
}

static __device__ __forceinline__ v8f wmma_bf16(v16bf a, v16bf b, v8f c) {
  // (neg_a, A, neg_b, B, c_mod, C, reuse_a, reuse_b)
  return __builtin_amdgcn_wmma_f32_16x16x32_bf16(false, a, false, b,
                                                 (short)0, c, false, false);
}

// A fragment (16x32, row-major along K). p = &row[ k + 8*halfgroup ].
// halves 0..7 -> K = 8h + i ; halves 8..15 -> K = 16 + 8h + i (ISA 7.12.2)
static __device__ __forceinline__ v16bf ld_a_frag(const bf16_t* p) {
  v8bf lo = *reinterpret_cast<const v8bf*>(p);
  v8bf hi = *reinterpret_cast<const v8bf*>(p + 16);
  v16bf r;
#pragma unroll
  for (int i = 0; i < 8; ++i) { r[i] = lo[i]; r[i + 8] = hi[i]; }
  return r;
}

// B fragment (32x16): lane = column, K = 8h + i (0..7), 16 + 8h + i (8..15).
// p = &rowOfTransposedB[ k + 8*halfgroup ].
static __device__ __forceinline__ v16bf ld_b_frag(const bf16_t* p) {
  v8bf lo = *reinterpret_cast<const v8bf*>(p);
  v8bf hi = *reinterpret_cast<const v8bf*>(p + 16);
  v16bf r;
#pragma unroll
  for (int i = 0; i < 8; ++i) { r[i] = lo[i]; r[i + 8] = hi[i]; }
  return r;
}

// Row reductions across the 16-lane half-wave (xor<16 stays in the half).
static __device__ __forceinline__ float redmax16(float v) {
  v = fmaxf(v, __shfl_xor(v, 1, 32));
  v = fmaxf(v, __shfl_xor(v, 2, 32));
  v = fmaxf(v, __shfl_xor(v, 4, 32));
  v = fmaxf(v, __shfl_xor(v, 8, 32));
  return v;
}
static __device__ __forceinline__ float redsum16(float v) {
  v += __shfl_xor(v, 1, 32);
  v += __shfl_xor(v, 2, 32);
  v += __shfl_xor(v, 4, 32);
  v += __shfl_xor(v, 8, 32);
  return v;
}

// Async copy of one 16B chunk per lane: global -> LDS (ASYNCcnt tracked).
static __device__ __forceinline__ void async_copy_b128(unsigned lds_off,
                                                       const bf16_t* gaddr) {
  asm volatile("global_load_async_to_lds_b128 %0, %1, off"
               :: "v"(lds_off), "v"(gaddr) : "memory");
}

// ---------------------------------------------------------------------------
// fp32 -> bf16 elementwise convert (vector4)
// ---------------------------------------------------------------------------
__global__ __launch_bounds__(256) void cvt_f32_bf16(const float* __restrict__ in,
                                                    bf16_t* __restrict__ out,
                                                    int n4) {
  int i = blockIdx.x * blockDim.x + threadIdx.x;
  if (i >= n4) return;
  v4f v = *reinterpret_cast<const v4f*>(in + 4 * (size_t)i);
  v4bf o;
#pragma unroll
  for (int e = 0; e < 4; ++e) o[e] = (bf16_t)v[e];
  *reinterpret_cast<v4bf*>(out + 4 * (size_t)i) = o;
}

// ---------------------------------------------------------------------------
// Weight transpose + convert: W [K=2048][N=2048] f32  ->  Wt [N][K] bf16
// ---------------------------------------------------------------------------
__global__ __launch_bounds__(256) void transpose_w_kernel(const float* __restrict__ W,
                                                          bf16_t* __restrict__ Wt) {
  __shared__ float tile[32][33];
  const int tx = threadIdx.x, ty = threadIdx.y;       // (32,8)
  const int n0 = blockIdx.x * 32, k0 = blockIdx.y * 32;
#pragma unroll
  for (int i = 0; i < 4; ++i)
    tile[ty + 8 * i][tx] = W[(size_t)(k0 + ty + 8 * i) * HID_C + n0 + tx];
  __syncthreads();
#pragma unroll
  for (int i = 0; i < 4; ++i)
    Wt[(size_t)(n0 + ty + 8 * i) * HID_C + k0 + tx] = (bf16_t)tile[tx][ty + 8 * i];
}

// ---------------------------------------------------------------------------
// Per-head V transpose: Vh [B*L][NH*HD] bf16 -> Vt [B][H][HD][L] bf16
// ---------------------------------------------------------------------------
__global__ __launch_bounds__(256) void transpose_v_kernel(const bf16_t* __restrict__ Vh,
                                                          bf16_t* __restrict__ Vt) {
  __shared__ bf16_t tile[32][33];
  const int tx = threadIdx.x, ty = threadIdx.y;       // (32,8)
  const int d0 = blockIdx.x * 32;                     // head-dim tile
  const int l0 = blockIdx.y * 32;                     // seq tile
  const int bh = blockIdx.z;                          // b*16 + h
  const int b = bh >> 4, h = bh & 15;
#pragma unroll
  for (int i = 0; i < 4; ++i)
    tile[ty + 8 * i][tx] =
        Vh[(size_t)(b * L_C + l0 + ty + 8 * i) * HID_C + h * HD_C + d0 + tx];
  __syncthreads();
#pragma unroll
  for (int i = 0; i < 4; ++i)
    Vt[((size_t)bh * HD_C + d0 + ty + 8 * i) * L_C + l0 + tx] = tile[tx][ty + 8 * i];
}

// ---------------------------------------------------------------------------
// RoPE in place on bf16 Q and K. 1/sqrt(HD) folded into Q.
// ---------------------------------------------------------------------------
__global__ __launch_bounds__(256) void rope_kernel(bf16_t* __restrict__ Q,
                                                   bf16_t* __restrict__ K) {
  const int i = blockIdx.x * blockDim.x + threadIdx.x;  // < TOK*NH*64
  const int d = i & 63;
  const int h = (i >> 6) & (NH_C - 1);
  const int t = i >> 10;
  const int pos = t & (L_C - 1);
  const float freq = __expf(-(float)d * 0.17988946f);   // ln(1e5)/64
  const float theta = (float)pos * 0.25f * freq;        // pos / ROPE_SCALE
  const float c = cosf(theta), s = sinf(theta);
  const size_t base = (size_t)t * HID_C + h * HD_C + d;
  const float sc = 0.08838834764831845f;                // 1/sqrt(128)
  float q1 = (float)Q[base], q2 = (float)Q[base + 64];
  Q[base]      = (bf16_t)((q1 * c - q2 * s) * sc);
  Q[base + 64] = (bf16_t)((q1 * s + q2 * c) * sc);
  float k1 = (float)K[base], k2 = (float)K[base + 64];
  K[base]      = (bf16_t)(k1 * c - k2 * s);
  K[base + 64] = (bf16_t)(k1 * s + k2 * c);
}

// ---------------------------------------------------------------------------
// bf16 GEMM: C[M,N] = A[M,K] * Bt^T (Bt[N][K]).
// Block = 4 waves = 128x128 C tile; K-step 32.
// Block tiles are staged in LDS via async global->LDS copies (double
// buffered, ASYNCcnt); each wave computes a 64x64 tile (16 wmma / step).
// LDS rows padded to 40 bf16 (20 dwords) -> conflict-free 16-lane reads.
// ---------------------------------------------------------------------------
template <bool F32OUT>
__global__ __launch_bounds__(128) void gemm_wmma_kernel(const bf16_t* __restrict__ A,
                                                        const bf16_t* __restrict__ Bt,
                                                        void* __restrict__ Cout,
                                                        int N, int K) {
  __shared__ __align__(16) bf16_t smA[2][128][40];
  __shared__ __align__(16) bf16_t smB[2][128][40];

  const int tid  = threadIdx.x;
  const int wave = tid >> 5;
  const int lane = tid & 31;
  const int half = lane >> 4;
  const int ln   = lane & 15;
  const int m0 = blockIdx.y * 128;
  const int n0 = blockIdx.x * 128;
  const int wm = (wave >> 1) * 64;            // wave tile inside block tile
  const int wn = (wave & 1) * 64;

  const int chunk = tid & 3;                  // 16B chunk within a 64B row
  const int rrow  = tid >> 2;                 // 0..31 (rows per async group)

  // Issue 8 async b128 instructions staging A[128][32] and B[128][32].
  auto stage = [&](int buf, int k) {
    const bf16_t* ga = A  + (size_t)(m0 + rrow) * K + k + chunk * 8;
    const bf16_t* gb = Bt + (size_t)(n0 + rrow) * K + k + chunk * 8;
#pragma unroll
    for (int t = 0; t < 4; ++t) {
      async_copy_b128((unsigned)(uintptr_t)(&smA[buf][rrow + 32 * t][chunk * 8]),
                      ga + (size_t)(32 * t) * K);
      async_copy_b128((unsigned)(uintptr_t)(&smB[buf][rrow + 32 * t][chunk * 8]),
                      gb + (size_t)(32 * t) * K);
    }
  };

  v8f acc[4][4];
#pragma unroll
  for (int i = 0; i < 4; ++i)
#pragma unroll
    for (int j = 0; j < 4; ++j) acc[i][j] = vzero8();

  stage(0, 0);

  for (int k = 0; k < K; k += 32) {
    const int buf = (k >> 5) & 1;
    if (k + 32 < K) {
      stage(buf ^ 1, k + 32);                         // prefetch next tile
      asm volatile("s_wait_asynccnt 0x8" ::: "memory");  // current tile done
    } else {
      asm volatile("s_wait_asynccnt 0x0" ::: "memory");
    }
    __syncthreads();                                  // tile visible to block

    v16bf af[4], bfr[4];
#pragma unroll
    for (int i = 0; i < 4; ++i)
      af[i] = ld_a_frag(&smA[buf][wm + 16 * i + ln][8 * half]);
#pragma unroll
    for (int j = 0; j < 4; ++j)
      bfr[j] = ld_b_frag(&smB[buf][wn + 16 * j + ln][8 * half]);
#pragma unroll
    for (int i = 0; i < 4; ++i)
#pragma unroll
      for (int j = 0; j < 4; ++j)
        acc[i][j] = wmma_bf16(af[i], bfr[j], acc[i][j]);

    __syncthreads();                                  // reads done before refill
  }

#pragma unroll
  for (int i = 0; i < 4; ++i)
#pragma unroll
    for (int j = 0; j < 4; ++j)
#pragma unroll
      for (int r = 0; r < 8; ++r) {
        const int row = m0 + wm + 16 * i + r + 8 * half;
        const int col = n0 + wn + 16 * j + ln;
        const float v = acc[i][j][r];
        if (F32OUT)
          reinterpret_cast<float*>(Cout)[(size_t)row * N + col] = v;
        else
          reinterpret_cast<bf16_t*>(Cout)[(size_t)row * N + col] = (bf16_t)v;
      }
}

// ---------------------------------------------------------------------------
// Flash-style causal attention. One wave per 16-query tile; 32 keys / iter.
// Qh,Kh: [B*L][NH*HD] bf16 (RoPE'd, Q pre-scaled). Vt: [B][H][HD][L] bf16.
// Oh: [B*L][NH*HD] bf16.
// ---------------------------------------------------------------------------
__global__ __launch_bounds__(128) void attn_kernel(const bf16_t* __restrict__ Qh,
                                                   const bf16_t* __restrict__ Kh,
                                                   const bf16_t* __restrict__ Vt,
                                                   bf16_t* __restrict__ Oh) {
  __shared__ __align__(16) float pbuf[4][16][36];   // per-wave P tile (16 q x 32 k)
  const int w    = threadIdx.x >> 5;
  const int lane = threadIdx.x & 31;
  const int half = lane >> 4;
  const int ln   = lane & 15;
  const int tile = blockIdx.x * 4 + w;              // (b*NH + h)*128 + qt
  const int qt = tile & 127;
  const int h  = (tile >> 7) & 15;
  const int b  = tile >> 11;

  const bf16_t* Qp = Qh + (size_t)b * L_C * HID_C + h * HD_C;
  const bf16_t* Kp = Kh + (size_t)b * L_C * HID_C + h * HD_C;
  const bf16_t* Vp = Vt + (size_t)(b * NH_C + h) * HD_C * L_C;

  // Q A-fragments over d = 0..127 (4 chunks of 32), resident in registers.
  v16bf qf[4];
#pragma unroll
  for (int c = 0; c < 4; ++c)
    qf[c] = ld_a_frag(Qp + (size_t)(qt * 16 + ln) * HID_C + 32 * c + 8 * half);

  v8f oacc[8];
  float rmax[8], rsum[8];
#pragma unroll
  for (int r = 0; r < 8; ++r) { rmax[r] = -3.0e38f; rsum[r] = 0.0f; }
#pragma unroll
  for (int j = 0; j < 8; ++j) oacc[j] = vzero8();

  float (*pb)[36] = pbuf[w];
  const int kbMax = (qt * 16 + 15) >> 5;            // causal: keys <= q row
  for (int kb = 0; kb <= kbMax; ++kb) {
    const int key0 = kb * 32;

    // ---- S = Q * K^T (two 16x16 tiles, K-dim = 128) ----
    v8f s0 = vzero8(), s1 = vzero8();
#pragma unroll
    for (int c = 0; c < 4; ++c) {
      v16bf b0 = ld_b_frag(Kp + (size_t)(key0 + ln)      * HID_C + 32 * c + 8 * half);
      v16bf b1 = ld_b_frag(Kp + (size_t)(key0 + 16 + ln) * HID_C + 32 * c + 8 * half);
      s0 = wmma_bf16(qf[c], b0, s0);
      s1 = wmma_bf16(qf[c], b1, s1);
    }

    // ---- causal mask + online softmax (row stats across 16-lane halves) ----
#pragma unroll
    for (int r = 0; r < 8; ++r) {
      const int qrow = qt * 16 + r + 8 * half;
      if (key0 + ln > qrow)      s0[r] += NEGV;
      if (key0 + 16 + ln > qrow) s1[r] += NEGV;
      float tm = redmax16(fmaxf(s0[r], s1[r]));
      float nm = fmaxf(rmax[r], tm);
      float corr = __expf(rmax[r] - nm);
      rmax[r] = nm;
      float p0 = __expf(s0[r] - nm);
      float p1 = __expf(s1[r] - nm);
      float ts = redsum16(p0 + p1);
      rsum[r] = rsum[r] * corr + ts;
#pragma unroll
      for (int j = 0; j < 8; ++j) oacc[j][r] *= corr;
      // stash P (C-layout) to wave-private LDS for A-layout reload
      pb[r + 8 * half][ln]      = p0;
      pb[r + 8 * half][16 + ln] = p1;
    }

    asm volatile("s_wait_dscnt 0" ::: "memory");    // C-layout stores -> A loads

    // ---- reload P as an A fragment (16x32 over keys) ----
    v16bf pa;
    {
      const float* prow = pb[ln];
      v4f a0 = *reinterpret_cast<const v4f*>(prow + 8 * half);
      v4f a1 = *reinterpret_cast<const v4f*>(prow + 8 * half + 4);
      v4f a2 = *reinterpret_cast<const v4f*>(prow + 16 + 8 * half);
      v4f a3 = *reinterpret_cast<const v4f*>(prow + 16 + 8 * half + 4);
#pragma unroll
      for (int i = 0; i < 4; ++i) {
        pa[i]      = (bf16_t)a0[i];
        pa[4 + i]  = (bf16_t)a1[i];
        pa[8 + i]  = (bf16_t)a2[i];
        pa[12 + i] = (bf16_t)a3[i];
      }
    }
    asm volatile("s_wait_dscnt 0" ::: "memory");    // finish reads before next stores

    // ---- O += P * V (8 column tiles of 16 over d=128) ----
#pragma unroll
    for (int j = 0; j < 8; ++j) {
      v16bf vf = ld_b_frag(Vp + (size_t)(16 * j + ln) * L_C + key0 + 8 * half);
      oacc[j] = wmma_bf16(pa, vf, oacc[j]);
    }
  }

  // ---- normalize and store ----
  const size_t orow0 = (size_t)b * L_C + (size_t)qt * 16;
#pragma unroll
  for (int j = 0; j < 8; ++j)
#pragma unroll
    for (int r = 0; r < 8; ++r) {
      const float v = oacc[j][r] / rsum[r];
      Oh[(orow0 + r + 8 * half) * HID_C + h * HD_C + 16 * j + ln] = (bf16_t)v;
    }
}

// ---------------------------------------------------------------------------
// Host-side orchestration
// ---------------------------------------------------------------------------
extern "C" void kernel_launch(void* const* d_in, const int* in_sizes, int n_in,
                              void* d_out, int out_size, void* d_ws, size_t ws_size,
                              hipStream_t stream) {
  (void)in_sizes; (void)n_in; (void)out_size; (void)ws_size;
  const float* x  = (const float*)d_in[0];
  // d_in[1] = mask: causal mask implemented analytically, unused.
  const float* Wq = (const float*)d_in[2];
  const float* Wk = (const float*)d_in[3];
  const float* Wv = (const float*)d_in[4];
  const float* Wo = (const float*)d_in[5];

  const size_t SZ_X = (size_t)TOK_C * HID_C;   // 8,388,608 elems
  const size_t SZ_W = (size_t)HID_C * HID_C;   // 4,194,304 elems

  bf16_t* xh  = (bf16_t*)d_ws;                 // x in bf16; reused as Oh later
  bf16_t* Wqt = xh  + SZ_X;
  bf16_t* Wkt = Wqt + SZ_W;
  bf16_t* Wvt = Wkt + SZ_W;
  bf16_t* Wot = Wvt + SZ_W;
  bf16_t* Qh  = Wot + SZ_W;
  bf16_t* Kh  = Qh  + SZ_X;
  bf16_t* Vh  = Kh  + SZ_X;
  bf16_t* Vt  = Vh  + SZ_X;
  bf16_t* Oh  = xh;                            // xh dead after QKV GEMMs

  // 1. x -> bf16
  {
    const int n4 = (int)(SZ_X / 4);
    cvt_f32_bf16<<<(n4 + 255) / 256, 256, 0, stream>>>(x, xh, n4);
  }

  // 2. Transpose + convert weights (W[K][N] f32 -> Wt[N][K] bf16)
  {
    dim3 tb(32, 8), tg(HID_C / 32, HID_C / 32);
    transpose_w_kernel<<<tg, tb, 0, stream>>>(Wq, Wqt);
    transpose_w_kernel<<<tg, tb, 0, stream>>>(Wk, Wkt);
    transpose_w_kernel<<<tg, tb, 0, stream>>>(Wv, Wvt);
    transpose_w_kernel<<<tg, tb, 0, stream>>>(Wo, Wot);
  }

  // 3. QKV projections (bf16 WMMA, async-LDS staged, bf16 out)
  {
    dim3 gg(HID_C / 128, TOK_C / 128);
    gemm_wmma_kernel<false><<<gg, 128, 0, stream>>>(xh, Wqt, Qh, HID_C, HID_C);
    gemm_wmma_kernel<false><<<gg, 128, 0, stream>>>(xh, Wkt, Kh, HID_C, HID_C);
    gemm_wmma_kernel<false><<<gg, 128, 0, stream>>>(xh, Wvt, Vh, HID_C, HID_C);
  }

  // 4. RoPE in place on Q (with 1/sqrt(HD)) and K
  rope_kernel<<<(TOK_C * NH_C * 64) / 256, 256, 0, stream>>>(Qh, Kh);

  // 5. Per-head V transpose for K-major B fragments
  transpose_v_kernel<<<dim3(HD_C / 32, L_C / 32, B_C * NH_C), dim3(32, 8), 0, stream>>>(Vh, Vt);

  // 6. Flash attention (one wave per 16-query tile; 4096 waves)
  attn_kernel<<<(B_C * NH_C * (L_C / 16)) / 4, 128, 0, stream>>>(Qh, Kh, Vt, Oh);

  // 7. Output projection -> fp32 d_out
  {
    dim3 gg(HID_C / 128, TOK_C / 128);
    gemm_wmma_kernel<true><<<gg, 128, 0, stream>>>(Oh, Wot, d_out, TOK_C, HID_C);
  }
}
